// RNNSequenceEncoder_21973052686874
// MI455X (gfx1250) — matrix operand order repaired and checked by reading, not compiled, not run on hardware
//
#include <hip/hip_runtime.h>
#include <hip/hip_bf16.h>

typedef __bf16 bf16_t;
typedef __attribute__((ext_vector_type(16))) __bf16 v16bf;
typedef __attribute__((ext_vector_type(8)))  __bf16 v8bf;
typedef __attribute__((ext_vector_type(8)))  float  v8f;
typedef __attribute__((ext_vector_type(4)))  float  v4f;

#define B_  256
#define T_  256
#define IN_ 128
#define S_  1024

// ---------------------------------------------------------------------------
// One-time prep kernels
// ---------------------------------------------------------------------------

__global__ void cvt_bf16(const float* __restrict__ src, bf16_t* __restrict__ dst, int n) {
    int i = blockIdx.x * blockDim.x + threadIdx.x;
    if (i < n) dst[i] = (bf16_t)src[i];
}

// W_rec fp32 [1024,1024] -> bf16 into Wcat1[:, 0:1024] (row stride 1152)
__global__ void cvt_wrec(const float* __restrict__ W, bf16_t* __restrict__ Wcat, int n) {
    int i = blockIdx.x * blockDim.x + threadIdx.x;
    if (i < n) {
        int r = i >> 10, k = i & 1023;
        Wcat[r * (S_ + IN_) + k] = (bf16_t)W[i];
    }
}

// W_comb = W_rec @ W_in -> bf16 into Wcat1[:, 1024:1152]  (fp32 precision fold)
__global__ void wcomb_kernel(const float* __restrict__ Wrec, const float* __restrict__ Win,
                             bf16_t* __restrict__ Wcat) {
    int n = blockIdx.x;
    int i = threadIdx.x;
    const float* wr = Wrec + n * S_;
    float s = 0.f;
    for (int k = 0; k < S_; ++k)
        s += wr[k] * Win[k * IN_ + i];
    Wcat[n * (S_ + IN_) + S_ + i] = (bf16_t)s;
}

// c[n] = b_rec[n] + sum_k W_rec[n][k] * b_in[k]
__global__ void cvec_kernel(const float* __restrict__ Wrec, const float* __restrict__ b_in,
                            const float* __restrict__ b_rec, float* __restrict__ c) {
    int n = blockIdx.x * blockDim.x + threadIdx.x;
    if (n < S_) {
        const float* wr = Wrec + n * S_;
        float s = b_rec[n];
        for (int k = 0; k < S_; ++k) s += wr[k] * b_in[k];
        c[n] = s;
    }
}

__global__ void zero_bf16(bf16_t* __restrict__ p, int n) {
    int i = blockIdx.x * blockDim.x + threadIdx.x;
    if (i < n) p[i] = (bf16_t)0.0f;
}

// ---------------------------------------------------------------------------
// Per-step WMMA GEMM:  Out = relu( [state | x_t] @ W^T + bias )
//   Each wave: 16(M) x 64(N) output tile, 4 accumulators, A reused 4x/iter.
//   K loop split: clean state loop (K=kA) then x loop (K=kX) -> no per-iter
//   address selects.  grid = 32 blocks x 256 threads = 256 waves.
// ---------------------------------------------------------------------------
__global__ __launch_bounds__(256)
void wmma_step(const bf16_t* __restrict__ Abase,   // state [256, kA]
               const bf16_t* __restrict__ Xbase,   // x_t slice (row stride xstride), or null
               const bf16_t* __restrict__ W,       // [1024, wstride]
               const float*  __restrict__ bias,    // [1024] fp32
               bf16_t* __restrict__ Obf,           // [256, 1024] bf16
               float*  __restrict__ Of32,          // optional fp32 copy (may be null)
               int kA, int kX, int wstride, int xstride) {
    const int lane = threadIdx.x & 31;
    const int wave = threadIdx.x >> 5;
    const int w    = blockIdx.x * 8 + wave;   // 256 waves
    const int m0   = (w & 15) * 16;           // 16 M tiles
    const int n0   = (w >> 4) * 64;           // 16 N groups of 64
    const int half = lane >> 4;               // 0: lanes 0-15, 1: lanes 16-31
    const int mrow = m0 + (lane & 15);
    const int ncol = lane & 15;

    v8f acc[4] = {v8f{}, v8f{}, v8f{}, v8f{}};

    const bf16_t* wr[4];
#pragma unroll
    for (int j = 0; j < 4; ++j)
        wr[j] = W + (n0 + j * 16 + ncol) * wstride + half * 16;

    // ---- state part: K = kA ----
    {
        const bf16_t* arow = Abase + mrow * kA + half * 16;
        for (int kb = 0; kb < kA; kb += 32) {
            v8bf lo = *(const v8bf*)(arow);
            v8bf hi = *(const v8bf*)(arow + 8);
            arow += 32;
            v16bf a;
#pragma unroll
            for (int i = 0; i < 8; ++i) { a[i] = lo[i]; a[i + 8] = hi[i]; }
#pragma unroll
            for (int j = 0; j < 4; ++j) {
                v16bf b = *(const v16bf*)(wr[j] + kb);
                acc[j] = __builtin_amdgcn_wmma_f32_16x16x32_bf16(
                    false, a, false, b, (short)0, acc[j], false, false);
            }
        }
    }

    // ---- x part: K = kX (folded input projection) ----
    if (kX > 0) {
        const bf16_t* xrow = Xbase + mrow * xstride + half * 16;
        for (int kb = 0; kb < kX; kb += 32) {
            v8bf lo = *(const v8bf*)(xrow + kb);
            v8bf hi = *(const v8bf*)(xrow + kb + 8);
            v16bf a;
#pragma unroll
            for (int i = 0; i < 8; ++i) { a[i] = lo[i]; a[i + 8] = hi[i]; }
#pragma unroll
            for (int j = 0; j < 4; ++j) {
                v16bf b = *(const v16bf*)(wr[j] + kA + kb);
                acc[j] = __builtin_amdgcn_wmma_f32_16x16x32_bf16(
                    false, a, false, b, (short)0, acc[j], false, false);
            }
        }
    }

    // ---- epilogue: bias + ReLU, bf16 store (+optional fp32 state copy) ----
    const bool wf32 = (Of32 != nullptr);
#pragma unroll
    for (int j = 0; j < 4; ++j) {
        const int nc = n0 + j * 16 + ncol;
        const float bj = bias[nc];
#pragma unroll
        for (int r = 0; r < 8; ++r) {
            const int m = m0 + half * 8 + r;
            float v = acc[j][r] + bj;
            v = v > 0.f ? v : 0.f;
            Obf[m * S_ + nc] = (bf16_t)v;
            if (wf32) Of32[m * S_ + nc] = v;
        }
    }
}

// ---------------------------------------------------------------------------
// Broadcast final state [B,S] fp32 into out [B,T,S] (vectorized float4)
// ---------------------------------------------------------------------------
__global__ void broadcast_out(const float* __restrict__ st, float* __restrict__ out) {
    unsigned idx = blockIdx.x * blockDim.x + threadIdx.x;   // over B*T*S/4 = 16777216
    unsigned s4 = idx & 255;           // S/4 = 256
    unsigned b  = idx >> 16;           // / (T * S/4)
    v4f v = *(const v4f*)(st + (size_t)b * S_ + s4 * 4);
    *(v4f*)(out + (size_t)idx * 4) = v;
}

// ---------------------------------------------------------------------------
// Host launcher
// ---------------------------------------------------------------------------
extern "C" void kernel_launch(void* const* d_in, const int* in_sizes, int n_in,
                              void* d_out, int out_size, void* d_ws, size_t ws_size,
                              hipStream_t stream) {
    const float* x      = (const float*)d_in[0];   // [B,T,IN]
    const float* W_in   = (const float*)d_in[1];   // [S,IN]
    const float* b_in   = (const float*)d_in[2];   // [S]
    const float* W_rec  = (const float*)d_in[3];   // [S,S]
    const float* b_rec  = (const float*)d_in[4];   // [S]
    const float* W_rec2 = (const float*)d_in[5];   // [S,S]
    const float* b_rec2 = (const float*)d_in[6];   // [S]
    float* out = (float*)d_out;

    // workspace partition (~23 MB, all 32B aligned)
    char* ws = (char*)d_ws;
    bf16_t* x_bf  = (bf16_t*)ws;  ws += (size_t)B_ * T_ * IN_ * 2;     // 16.8 MB
    bf16_t* Wcat1 = (bf16_t*)ws;  ws += (size_t)S_ * (S_ + IN_) * 2;   // 2.36 MB (W_rec | W_comb)
    bf16_t* Wr2bf = (bf16_t*)ws;  ws += (size_t)S_ * S_ * 2;           // 2.10 MB
    float*  cvec  = (float*)ws;   ws += (size_t)S_ * 4;                // fused bias for GEMM1
    bf16_t* state = (bf16_t*)ws;  ws += (size_t)B_ * S_ * 2;
    bf16_t* h1    = (bf16_t*)ws;  ws += (size_t)B_ * S_ * 2;
    float*  statef= (float*)ws;   ws += (size_t)B_ * S_ * 4;

    // ---- one-time prep ----
    cvt_bf16<<<(B_ * T_ * IN_ + 255) / 256, 256, 0, stream>>>(x, x_bf, B_ * T_ * IN_);
    cvt_wrec<<<(S_ * S_ + 255) / 256, 256, 0, stream>>>(W_rec, Wcat1, S_ * S_);
    cvt_bf16<<<(S_ * S_ + 255) / 256, 256, 0, stream>>>(W_rec2, Wr2bf, S_ * S_);
    wcomb_kernel<<<S_, IN_, 0, stream>>>(W_rec, W_in, Wcat1);
    cvec_kernel<<<(S_ + 255) / 256, 256, 0, stream>>>(W_rec, b_in, b_rec, cvec);
    zero_bf16<<<(B_ * S_ + 255) / 256, 256, 0, stream>>>(state, B_ * S_);

    // ---- sequential scan: 2 WMMA GEMMs per timestep ----
    for (int t = 0; t < T_; ++t) {
        // h1 = relu(state @ W_rec^T + x_t @ W_comb^T + c)   [K = 1024 + 128]
        wmma_step<<<32, 256, 0, stream>>>(state, x_bf + t * IN_, Wcat1, cvec,
                                          h1, nullptr, S_, IN_, S_ + IN_, T_ * IN_);
        // state = relu(h1 @ W_rec2^T + b_rec2)               [K = 1024]
        wmma_step<<<32, 256, 0, stream>>>(h1, nullptr, Wr2bf, b_rec2,
                                          state, statef, S_, 0, S_, 0);
    }

    // ---- broadcast final state into out[B,T,S] ----
    broadcast_out<<<(B_ * T_ * S_ / 4) / 256, 256, 0, stream>>>(statef, out);
}
